// Block_81501299409411
// MI455X (gfx1250) — matrix operand order repaired and checked by reading, not compiled, not run on hardware
//
#include <hip/hip_runtime.h>
#include <cstdint>
#include <cstddef>

// ---------- types ----------
typedef __bf16        v16bf  __attribute__((ext_vector_type(16)));
typedef float         v8f    __attribute__((ext_vector_type(8)));
typedef unsigned int  u32x4  __attribute__((ext_vector_type(4)));
typedef unsigned int  u32x2  __attribute__((ext_vector_type(2)));
typedef unsigned short u16;

#define WMMA_BF16 __builtin_amdgcn_wmma_f32_16x16x32_bf16

union V16U { v16bf v; u32x4 q[2]; };

__device__ __forceinline__ u16 f2bf(float f) {
  unsigned u = __float_as_uint(f);
  u += 0x7FFFu + ((u >> 16) & 1u);          // round-to-nearest-even
  return (u16)(u >> 16);
}

// ---------- fp32 -> bf16 elementwise (weights) ----------
__global__ void cvt_f32_bf16(const float* __restrict__ src, u16* __restrict__ dst, int n) {
  int i = blockIdx.x * blockDim.x + threadIdx.x;
  if (i < n) dst[i] = f2bf(src[i]);
}

// ---------- LayerNorm: one wave per 512-elem row, bf16 out ----------
__global__ void layernorm_bf16(const float* __restrict__ x, const float* __restrict__ w,
                               const float* __restrict__ bta, u16* __restrict__ out, int rows) {
  const int C = 512;
  int lane = threadIdx.x & 31;
  int row  = blockIdx.x * (blockDim.x >> 5) + (threadIdx.x >> 5);
  if (row >= rows) return;
  const float* xr = x + (size_t)row * C;
  float4 v[4];
  float s = 0.f, s2 = 0.f;
#pragma unroll
  for (int i = 0; i < 4; ++i) {
    v[i] = *(const float4*)(xr + (i * 32 + lane) * 4);
    s  += v[i].x + v[i].y + v[i].z + v[i].w;
    s2 += v[i].x*v[i].x + v[i].y*v[i].y + v[i].z*v[i].z + v[i].w*v[i].w;
  }
#pragma unroll
  for (int m = 16; m >= 1; m >>= 1) { s += __shfl_xor(s, m); s2 += __shfl_xor(s2, m); }
  float mu   = s  * (1.0f / 512.0f);
  float var  = s2 * (1.0f / 512.0f) - mu * mu;
  float rstd = rsqrtf(var + 1e-5f);
  u16* orow = out + (size_t)row * C;
#pragma unroll
  for (int i = 0; i < 4; ++i) {
    int c = (i * 32 + lane) * 4;
    float4 wv = *(const float4*)(w + c);
    float4 bv = *(const float4*)(bta + c);
    float y0 = (v[i].x - mu) * rstd * wv.x + bv.x;
    float y1 = (v[i].y - mu) * rstd * wv.y + bv.y;
    float y2 = (v[i].z - mu) * rstd * wv.z + bv.z;
    float y3 = (v[i].w - mu) * rstd * wv.w + bv.w;
    u32x2 pk;
    pk.x = (unsigned)f2bf(y0) | ((unsigned)f2bf(y1) << 16);
    pk.y = (unsigned)f2bf(y2) | ((unsigned)f2bf(y3) << 16);
    *(u32x2*)(orow + c) = pk;
  }
}

// ---------- bf16 WMMA GEMM with async-LDS A-panel staging ----------
// Block = 8 waves; block owns rows m0..m0+15 and a 512-col slab (wave w -> 64 cols).
// A panel (16 x panelK) staged global->LDS via global_load_async_to_lds_b128,
// consumed via ds_load_b128; B double-buffered in registers from global/L2.
// flags: 1 = GELU, 2 = add residual (fp32, stride N), 4 = fp32 output (else bf16)
__global__ void gemm_wmma(const u16* __restrict__ A, const u16* __restrict__ W,
                          const float* __restrict__ bias, const float* __restrict__ res,
                          float* __restrict__ outF, u16* __restrict__ outB,
                          int M, int N, int K, int flags) {
  extern __shared__ u16 smem[];
  const int lane = threadIdx.x & 31;
  const int l16  = lane & 15;
  const int half = lane >> 4;
  const int wav  = threadIdx.x >> 5;

  const int colBlocks = N >> 9;                       // 512-col slabs
  const int m0 = (blockIdx.x / colBlocks) << 4;
  const int n0 = (blockIdx.x % colBlocks) * 512 + wav * 64;

  const int panelK    = (K > 1024) ? 1024 : K;
  const int ldsStride = panelK + 8;                   // +16B pad rotates banks per row
  const unsigned sbase = (unsigned)(size_t)(&smem[0]);

  v8f zero = {0.f,0.f,0.f,0.f,0.f,0.f,0.f,0.f};
  v8f acc[4] = {zero, zero, zero, zero};

  V16U bufA[4], bufB[4];
  auto loadB = [&](V16U* bb, int kglob) {
#pragma unroll
    for (int t = 0; t < 4; ++t) {
      const u16* wRow = W + (size_t)(n0 + t * 16 + l16) * K + kglob + half * 8;
      bb[t].q[0] = *(const u32x4*)(wRow);
      bb[t].q[1] = *(const u32x4*)(wRow + 16);
    }
  };
  auto tileA = [&](int kloc) {
    V16U a;
    const u16* sp = smem + (size_t)l16 * ldsStride + kloc + half * 8;
    a.q[0] = *(const u32x4*)(sp);          // K 0..7  / 8..15   (ds_load_b128)
    a.q[1] = *(const u32x4*)(sp + 16);     // K 16..23/ 24..31
    return a;
  };

  const int cpr    = panelK >> 3;                     // 16B chunks per row
  const int cprLog = 31 - __clz(cpr);
  const int totalChunks = 16 * cpr;

  for (int p = 0; p < K; p += panelK) {
    __syncthreads();                                  // panel reuse fence
    // ---- async stage A panel: global -> LDS ----
    for (int c = threadIdx.x; c < totalChunks; c += 256) {
      int r   = c >> cprLog;
      int cwi = c & (cpr - 1);
      const u16* g = A + (size_t)(m0 + r) * K + p + cwi * 8;
      unsigned lds = sbase + (unsigned)(r * ldsStride + cwi * 8) * 2u;
      asm volatile("global_load_async_to_lds_b128 %0, %1, off"
                   :: "v"(lds), "v"(g) : "memory");
    }
    asm volatile("s_wait_asynccnt 0x0" ::: "memory");
    __syncthreads();

    // ---- k-loop: A from LDS, B double-buffered in registers ----
    loadB(bufA, p);
    for (int kk = 0; kk < panelK; kk += 64) {
      loadB(bufB, p + kk + 32);
      V16U a0 = tileA(kk);
#pragma unroll
      for (int t = 0; t < 4; ++t)
        acc[t] = WMMA_BF16(false, a0.v, false, bufA[t].v, (short)0, acc[t], false, false);
      if (kk + 64 < panelK) loadB(bufA, p + kk + 64);
      V16U a1 = tileA(kk + 32);
#pragma unroll
      for (int t = 0; t < 4; ++t)
        acc[t] = WMMA_BF16(false, a1.v, false, bufB[t].v, (short)0, acc[t], false, false);
    }
  }

  // ---- epilogue ----
  const bool doGelu = flags & 1, hasRes = flags & 2, f32out = flags & 4;
#pragma unroll
  for (int t = 0; t < 4; ++t) {
    int n = n0 + t * 16 + l16;
    float bv = bias[n];
#pragma unroll
    for (int j = 0; j < 8; ++j) {
      int m = m0 + j + half * 8;                 // C-layout: vgpr j, lane-half -> row
      float v = acc[t][j] + bv;
      if (doGelu) {
        float x3 = v * v * v;
        v = 0.5f * v * (1.0f + tanhf(0.7978845608028654f * (v + 0.044715f * x3)));
      }
      if (hasRes) v += res[(size_t)m * N + n];
      if (f32out) outF[(size_t)m * N + n] = v;
      else        outB[(size_t)m * N + n] = f2bf(v);
    }
  }
}

// ---------- V transpose: qkv[b,t,2C+h*32+d] -> vt[((b*H+h)*32+d)*T + t] ----------
__global__ void vtrans(const u16* __restrict__ qkv, u16* __restrict__ vt) {
  int idx = blockIdx.x * blockDim.x + threadIdx.x;     // 8*16*32*1024 = 4194304
  int t  = idx & 1023;
  int r  = idx >> 10;
  int d  = r & 31;
  int hb = r >> 5;
  int h  = hb & 15;
  int b  = hb >> 4;
  vt[idx] = qkv[((size_t)(b * 1024 + t)) * 1536 + 1024 + h * 32 + d];
}

// ---------- flash attention: 16 queries per wave, 32-key steps ----------
__global__ void attn_flash(const u16* __restrict__ qkv, const u16* __restrict__ vt,
                           u16* __restrict__ y) {
  const int T = 1024, C = 512, H = 16;
  const int lane = threadIdx.x & 31, l16 = lane & 15, half = lane >> 4;
  int wid = blockIdx.x * (blockDim.x >> 5) + (threadIdx.x >> 5);
  int b   = wid >> 10;                // H*(T/16) = 1024 waves per batch
  int rem = wid & 1023;
  int h   = rem >> 6;
  int t0  = (rem & 63) << 4;

  const u16* qk = qkv + (size_t)b * T * (3 * C);

  // Q^T as B operand: lane = query column, halves hold d {0..7,16..23} / {8..15,24..31}
  V16U qB;
  { const u16* p = qk + (size_t)(t0 + l16) * 1536 + h * 32 + half * 8;
    qB.q[0] = *(const u32x4*)p; qB.q[1] = *(const u32x4*)(p + 16); }

  v8f O0 = {0.f,0.f,0.f,0.f,0.f,0.f,0.f,0.f};
  v8f O1 = O0;
  float mrow = -3.0e38f, lrow = 0.0f;
  const float scale = 0.17677669529663689f;   // 1/sqrt(32)

  const u16* kbaseP = qk + C + h * 32;                       // K block base
  const u16* vtb    = vt + (((size_t)(b * H + h)) << 5) * T; // [32 d][1024 t]

  for (int kb = 0; kb < T; kb += 32) {
    // K chunk as A operand (rows = keys, d {0..7,16..23} / {8..15,24..31} per half)
    V16U kA0, kA1;
    { const u16* p = kbaseP + (size_t)(kb + l16) * 1536 + half * 8;
      kA0.q[0] = *(const u32x4*)p; kA0.q[1] = *(const u32x4*)(p + 16); }
    { const u16* p = kbaseP + (size_t)(kb + 16 + l16) * 1536 + half * 8;
      kA1.q[0] = *(const u32x4*)p; kA1.q[1] = *(const u32x4*)(p + 16); }

    v8f z = {0.f,0.f,0.f,0.f,0.f,0.f,0.f,0.f};
    v8f st0 = WMMA_BF16(false, kA0.v, false, qB.v, (short)0, z, false, false); // S^T keys 0..15
    v8f st1 = WMMA_BF16(false, kA1.v, false, qB.v, (short)0, z, false, false); // S^T keys 16..31

    // online softmax: lane = query, so row reduce = vgpr loop + xor16
    float pm = -3.0e38f;
#pragma unroll
    for (int j = 0; j < 8; ++j) {
      st0[j] *= scale; st1[j] *= scale;
      pm = fmaxf(pm, fmaxf(st0[j], st1[j]));
    }
    pm = fmaxf(pm, __shfl_xor(pm, 16));
    float mnew  = fmaxf(mrow, pm);
    float alpha = __expf(mrow - mnew);

    float rs = 0.0f;
    V16U pA;                                    // P^T C-layout == A-layout, no shuffle
#pragma unroll
    for (int j = 0; j < 8; ++j) {
      float p0 = __expf(st0[j] - mnew);
      float p1 = __expf(st1[j] - mnew);
      rs += p0 + p1;
      pA.v[j]     = (__bf16)p0;
      pA.v[8 + j] = (__bf16)p1;
    }
    rs += __shfl_xor(rs, 16);
    lrow = lrow * alpha + rs;
    mrow = mnew;

    // rescale O: per-row alpha broadcast (row = vgpr j + 8*half)
#pragma unroll
    for (int j = 0; j < 8; ++j) {
      float f = __shfl(alpha, j + half * 8);
      O0[j] *= f; O1[j] *= f;
    }

    // V chunk as B operand from transposed vt: lane = d column, keys contiguous
    V16U vB0, vB1;
    { const u16* p = vtb + (size_t)l16 * T + kb + half * 8;
      vB0.q[0] = *(const u32x4*)p; vB0.q[1] = *(const u32x4*)(p + 16); }
    { const u16* p = vtb + (size_t)(16 + l16) * T + kb + half * 8;
      vB1.q[0] = *(const u32x4*)p; vB1.q[1] = *(const u32x4*)(p + 16); }

    O0 = WMMA_BF16(false, pA.v, false, vB0.v, (short)0, O0, false, false);
    O1 = WMMA_BF16(false, pA.v, false, vB1.v, (short)0, O1, false, false);
  }

  float inv = 1.0f / lrow;
  u16* yb = y + (size_t)b * T * C;
#pragma unroll
  for (int j = 0; j < 8; ++j) {
    float f = __shfl(inv, j + half * 8);
    int t = t0 + j + half * 8;
    yb[(size_t)t * C + h * 32 + l16]      = f2bf(O0[j] * f);
    yb[(size_t)t * C + h * 32 + 16 + l16] = f2bf(O1[j] * f);
  }
}

// ---------- host-side orchestration ----------
extern "C" void kernel_launch(void* const* d_in, const int* in_sizes, int n_in,
                              void* d_out, int out_size, void* d_ws, size_t ws_size,
                              hipStream_t stream) {
  (void)in_sizes; (void)n_in; (void)out_size; (void)ws_size;
  const float* x       = (const float*)d_in[0];
  const float* ln1_w   = (const float*)d_in[1];
  const float* ln1_b   = (const float*)d_in[2];
  const float* attn_w  = (const float*)d_in[3];
  const float* attn_b  = (const float*)d_in[4];
  const float* proj_w  = (const float*)d_in[5];
  const float* proj_b  = (const float*)d_in[6];
  const float* ln2_w   = (const float*)d_in[7];
  const float* ln2_b   = (const float*)d_in[8];
  const float* fc_w    = (const float*)d_in[9];
  const float* fc_b    = (const float*)d_in[10];
  const float* cproj_w = (const float*)d_in[11];
  const float* cproj_b = (const float*)d_in[12];
  float* out = (float*)d_out;

  const int C = 512, T = 1024, Bt = 8, M = Bt * T;   // M = 8192

  char* base = (char*)d_ws;
  size_t off = 0;
  auto alloc = [&](size_t bytes) -> char* {
    char* p = base + off;
    off = (off + bytes + 255) & ~(size_t)255;
    return p;
  };
  // total ~94 MB of workspace (with y->xh and h2->vt aliasing)
  u16*  wqkv   = (u16*)alloc((size_t)3 * C * C * 2);
  u16*  wproj  = (u16*)alloc((size_t)C * C * 2);
  u16*  wfc    = (u16*)alloc((size_t)4 * C * C * 2);
  u16*  wcproj = (u16*)alloc((size_t)4 * C * C * 2);
  u16*  xh     = (u16*)alloc((size_t)M * C * 2);        // ln1 out; later reused as y
  u16*  qkvb   = (u16*)alloc((size_t)M * 3 * C * 2);
  u16*  vt     = (u16*)alloc((size_t)M * C * 2);        // V^T; later reused as h2
  u16*  ub     = (u16*)alloc((size_t)M * 4 * C * 2);
  float* x1    = (float*)alloc((size_t)M * C * 4);

  auto ldsBytes = [](int K) -> size_t {
    int panelK = (K > 1024) ? 1024 : K;
    return (size_t)16 * (panelK + 8) * 2;
  };

  // 1) weights to bf16
  cvt_f32_bf16<<<(3*C*C + 255)/256, 256, 0, stream>>>(attn_w,  wqkv,   3*C*C);
  cvt_f32_bf16<<<(C*C   + 255)/256, 256, 0, stream>>>(proj_w,  wproj,  C*C);
  cvt_f32_bf16<<<(4*C*C + 255)/256, 256, 0, stream>>>(fc_w,    wfc,    4*C*C);
  cvt_f32_bf16<<<(4*C*C + 255)/256, 256, 0, stream>>>(cproj_w, wcproj, 4*C*C);

  // 2) ln1
  layernorm_bf16<<<M/8, 256, 0, stream>>>(x, ln1_w, ln1_b, xh, M);
  // 3) qkv = xh @ attn_w^T + attn_b      [8192 x 1536], K=512
  gemm_wmma<<<(M/16)*(3*C/512), 256, ldsBytes(C), stream>>>(
      xh, wqkv, attn_b, nullptr, nullptr, qkvb, M, 3*C, C, 0);
  // 4) V transpose for attention B-operand
  vtrans<<<(Bt*16*32*1024)/256, 256, 0, stream>>>(qkvb, vt);
  // 5) flash attention -> y (reuses xh region)
  attn_flash<<<(Bt*16*(T/16))/8, 256, 0, stream>>>(qkvb, vt, xh);
  // 6) x1 = x + y @ proj_w^T + proj_b    (fp32 residual stream)
  gemm_wmma<<<(M/16)*(C/512), 256, ldsBytes(C), stream>>>(
      xh, wproj, proj_b, x, x1, nullptr, M, C, C, 2 | 4);
  // 7) ln2 -> h2 (reuses vt region)
  layernorm_bf16<<<M/8, 256, 0, stream>>>(x1, ln2_w, ln2_b, vt, M);
  // 8) u = gelu(h2 @ fc_w^T + fc_b)      [8192 x 2048], K=512
  gemm_wmma<<<(M/16)*(4*C/512), 256, ldsBytes(C), stream>>>(
      vt, wfc, fc_b, nullptr, nullptr, ub, M, 4*C, C, 1);
  // 9) out = x1 + u @ cproj_w^T + cproj_b  [8192 x 512], K=2048
  gemm_wmma<<<(M/16)*(C/512), 256, ldsBytes(4*C), stream>>>(
      ub, wcproj, cproj_b, x1, out, nullptr, M, C, 4*C, 2 | 4);
}
